// EnhancedAdaptiveMDM_8186207666403
// MI455X (gfx1250) — compile-verified, matrix operand order checked by reading, not run on hardware
//
#include <hip/hip_runtime.h>
#include <hip/hip_bf16.h>
#include <math.h>

// ---------------------------------------------------------------------------
// Problem constants (from reference): B=64, F=128, S=2048, K_LIST={8,4,2}
// ---------------------------------------------------------------------------
#define BB 64
#define FF 128
#define SS 2048
#define MM (BB * FF)            // 8192 rows for all GEMMs
#define FS (FF * SS)            // 262144 features for batchnorm
#define EPSV 1e-5f

typedef __attribute__((ext_vector_type(16))) __bf16 v16bf;
typedef __attribute__((ext_vector_type(8)))  __bf16 v8bf;
typedef __attribute__((ext_vector_type(8)))  float  v8f;

// ---------------------------------------------------------------------------
// 1) BatchNorm over batch axis. One thread per feature column.
//    Second pass re-reads x (67MB fits in 192MB L2).  Writes xn (ws) and
//    initializes d_out = xn  (samples[3] before the += update).
// ---------------------------------------------------------------------------
__global__ __launch_bounds__(256) void bn_kernel(const float* __restrict__ x,
                                                 const float* __restrict__ gamma,
                                                 const float* __restrict__ beta,
                                                 float* __restrict__ xn,
                                                 float* __restrict__ out) {
    int c = blockIdx.x * 256 + threadIdx.x;           // 0 .. FS-1
    float s = 0.f, q = 0.f;
    for (int b = 0; b < BB; ++b) {
        float v = x[b * FS + c];
        s += v; q += v * v;
    }
    float mean = s * (1.f / BB);
    float var  = q * (1.f / BB) - mean * mean;
    float g    = gamma[c] * rsqrtf(var + EPSV);
    float be   = beta[c];
    for (int b = 0; b < BB; ++b) {
        float v = (x[b * FS + c] - mean) * g + be;
        xn[b * FS + c]  = v;
        out[b * FS + c] = v;
    }
}

// ---------------------------------------------------------------------------
// Weight prep: wt(N x K) bf16 = transpose of w(K x N) f32.  One-time cost so
// the GEMM's B-tile fill becomes a pure K-contiguous copy (same as A).
// ---------------------------------------------------------------------------
__global__ __launch_bounds__(256) void wprep_kernel(const float* __restrict__ w,
                                                    __bf16* __restrict__ wt,
                                                    int K, int N) {
    int i = blockIdx.x * 256 + threadIdx.x;           // over N*K (output index)
    int n = i / K;
    int k = i - n * K;
    wt[i] = (__bf16)w[(long)k * N + n];
}

// ---------------------------------------------------------------------------
// 2) Pooling: avg & max over k contiguous elements.
// ---------------------------------------------------------------------------
__global__ __launch_bounds__(256) void pool_kernel(const float* __restrict__ xn,
                                                   float* __restrict__ avg,
                                                   float* __restrict__ mx,
                                                   int L, int k) {
    int i = blockIdx.x * 256 + threadIdx.x;           // 0 .. MM*L-1
    int m = i / L;
    int l = i - m * L;
    const float* p = xn + (long)m * SS + (long)l * k;
    float s = 0.f, mv = -3.4e38f;
    for (int j = 0; j < k; ++j) { float v = p[j]; s += v; mv = fmaxf(mv, v); }
    avg[i] = s / (float)k;
    mx[i]  = mv;
}

// ---------------------------------------------------------------------------
// 3) Per-(b,f) stats of pfs = 0.5*(avg+mx): mean, std(ddof=1), max.
// ---------------------------------------------------------------------------
__global__ __launch_bounds__(256) void stats_kernel(const float* __restrict__ avg,
                                                    const float* __restrict__ mx,
                                                    float* __restrict__ stats,
                                                    int L) {
    __shared__ float ss[256], sq[256], sm[256];
    int m = blockIdx.x;
    int b = m >> 7, f = m & 127;
    const float* pa = avg + (long)m * L;
    const float* pm = mx  + (long)m * L;
    float s = 0.f, q = 0.f, mv = -3.4e38f;
    for (int i = threadIdx.x; i < L; i += 256) {
        float p = 0.5f * (pa[i] + pm[i]);
        s += p; q += p * p; mv = fmaxf(mv, p);
    }
    ss[threadIdx.x] = s; sq[threadIdx.x] = q; sm[threadIdx.x] = mv;
    __syncthreads();
    for (int d = 128; d > 0; d >>= 1) {
        if (threadIdx.x < d) {
            ss[threadIdx.x] += ss[threadIdx.x + d];
            sq[threadIdx.x] += sq[threadIdx.x + d];
            sm[threadIdx.x]  = fmaxf(sm[threadIdx.x], sm[threadIdx.x + d]);
        }
        __syncthreads();
    }
    if (threadIdx.x == 0) {
        float mean = ss[0] / (float)L;
        float var  = (sq[0] - (float)L * mean * mean) / (float)(L - 1);
        stats[b * 384 + f]       = mean;
        stats[b * 384 + 128 + f] = sqrtf(fmaxf(var, 0.f));
        stats[b * 384 + 256 + f] = sm[0];
    }
}

// ---------------------------------------------------------------------------
// 4) Selector MLP: sigmoid(relu(stats@W1+b1)@W2+b2).  One block per batch row.
// ---------------------------------------------------------------------------
__global__ __launch_bounds__(64) void sel_kernel(const float* __restrict__ stats,
                                                 const float* __restrict__ w1,
                                                 const float* __restrict__ b1,
                                                 const float* __restrict__ w2,
                                                 const float* __restrict__ b2,
                                                 float* __restrict__ wsel) {
    __shared__ float red[64];
    int b = blockIdx.x;
    int j = threadIdx.x;                              // hidden unit 0..63
    float h = b1[j];
    const float* srow = stats + b * 384;
    for (int c = 0; c < 384; ++c) h += srow[c] * w1[c * 64 + j];
    red[j] = fmaxf(h, 0.f) * w2[j];
    __syncthreads();
    for (int d = 32; d > 0; d >>= 1) {
        if (j < d) red[j] += red[j + d];
        __syncthreads();
    }
    if (j == 0) {
        float logit = red[0] + b2[0];
        wsel[b] = 1.f / (1.f + expf(-logit));
    }
}

// ---------------------------------------------------------------------------
// 5) Mix: sample = w*avg + (1-w)*mx; f32 into avg (GEMM accumulator) and a
//    bf16 shadow copy (GEMM A-operand).
// ---------------------------------------------------------------------------
__global__ __launch_bounds__(256) void mix_kernel(float* __restrict__ avg,
                                                  const float* __restrict__ mx,
                                                  const float* __restrict__ wsel,
                                                  __bf16* __restrict__ sampleb,
                                                  int L) {
    int i = blockIdx.x * 256 + threadIdx.x;
    int m = i / L;
    float w = wsel[m >> 7];                           // b = m / 128
    float v = w * avg[i] + (1.f - w) * mx[i];
    avg[i]     = v;
    sampleb[i] = (__bf16)v;
}

// ---------------------------------------------------------------------------
// 6) WMMA GEMM (all operands bf16 in global, f32 accumulate):
//      tmp = A(bf16, MxK) @ Wt(bf16, NxK)^T + bias(N)
//    MODE 0: outb = (bf16)gelu_exact(tmp)                (hidden H, bf16-only)
//    MODE 1: out += tmp; if (outb) outb = (bf16)out      (residual + shadow)
//    128x128 block tile, 8 wave32 waves x (64x32 = 4x2 wmma tiles), K step 32.
//    Register-staged 2-stage pipeline: fetch (4x global_load_b128, no waits)
//    -> ds_load+8x v_wmma compute -> commit (wait + 4x ds_store_b128) ->
//    barrier.  Global latency hides behind the WMMA phase.
// ---------------------------------------------------------------------------
__device__ __forceinline__ float gelu_exact(float v) {
    return 0.5f * v * (1.0f + erff(v * 0.70710678118654752f));
}

#define LDT 40   // LDS row stride in bf16 (80B): 16B-aligned, bank-spread

template <int MODE>
__global__ __launch_bounds__(256) void gemm_kernel(const __bf16* __restrict__ A,
                                                   const __bf16* __restrict__ Wt,
                                                   const float* __restrict__ bias,
                                                   float* __restrict__ out,
                                                   __bf16* __restrict__ outb,
                                                   int N, int K) {
    __shared__ __align__(16) __bf16 As[2][128 * LDT];
    __shared__ __align__(16) __bf16 Bs[2][128 * LDT];

    const int tid    = threadIdx.x;
    const int m_base = blockIdx.y * 128;
    const int n_base = blockIdx.x * 128;
    const int wave   = tid >> 5;
    const int lane   = tid & 31;
    const int wm     = (wave >> 2) * 64;              // 0 or 64
    const int wn     = (wave & 3) * 32;               // 0,32,64,96
    const int r      = lane & 15;
    const int koff   = (lane >> 4) * 8;               // half-wave K split: 0 or 8

    // Tile copy mapping: 128 rows x 32 bf16 = 512 x 16B chunks, 2 per thread.
    // idx = tid + i*256: row = idx>>2 (0..127), chunk c = idx&3 (8 bf16 each).
    const int row0c = (tid + 0)   >> 2, c0 = ((tid + 0)   & 3) * 8;
    const int row1c = (tid + 256) >> 2, c1 = ((tid + 256) & 3) * 8;

    uint4 ra0, ra1, rb0, rb1;                         // staging registers

    auto fetch = [&](int k0) {
        ra0 = *(const uint4*)(A  + (long)(m_base + row0c) * K + k0 + c0);
        ra1 = *(const uint4*)(A  + (long)(m_base + row1c) * K + k0 + c1);
        rb0 = *(const uint4*)(Wt + (long)(n_base + row0c) * K + k0 + c0);
        rb1 = *(const uint4*)(Wt + (long)(n_base + row1c) * K + k0 + c1);
    };
    auto commit = [&](int buf) {
        *(uint4*)&As[buf][row0c * LDT + c0] = ra0;
        *(uint4*)&As[buf][row1c * LDT + c1] = ra1;
        *(uint4*)&Bs[buf][row0c * LDT + c0] = rb0;
        *(uint4*)&Bs[buf][row1c * LDT + c1] = rb1;
    };

    v8f acc[4][2] = {};

    fetch(0);
    commit(0);
    __syncthreads();

    const int nsteps = K >> 5;
    for (int s = 0; s < nsteps; ++s) {
        const int buf = s & 1;
        const bool more = (s + 1) < nsteps;
        if (more) fetch((s + 1) << 5);                // async: no wait yet

        v16bf a[4], b[2];
        #pragma unroll
        for (int j = 0; j < 2; ++j) {
            const __bf16* base = &Bs[buf][(wn + j * 16 + r) * LDT + koff];
            v8bf lo = *(const v8bf*)(base);
            v8bf hi = *(const v8bf*)(base + 16);
            b[j] = __builtin_shufflevector(lo, hi, 0, 1, 2, 3, 4, 5, 6, 7,
                                                   8, 9, 10, 11, 12, 13, 14, 15);
        }
        #pragma unroll
        for (int i = 0; i < 4; ++i) {
            const __bf16* base = &As[buf][(wm + i * 16 + r) * LDT + koff];
            v8bf lo = *(const v8bf*)(base);
            v8bf hi = *(const v8bf*)(base + 16);
            a[i] = __builtin_shufflevector(lo, hi, 0, 1, 2, 3, 4, 5, 6, 7,
                                                   8, 9, 10, 11, 12, 13, 14, 15);
        }
        #pragma unroll
        for (int i = 0; i < 4; ++i)
            #pragma unroll
            for (int j = 0; j < 2; ++j)
                acc[i][j] = __builtin_amdgcn_wmma_f32_16x16x32_bf16(
                    false, a[i], false, b[j], (short)0, acc[i][j], false, false);

        if (more) commit(buf ^ 1);                    // wait lands here, post-WMMA
        __syncthreads();
    }

    // Epilogue.  C/D layout: lane 0-15 -> N=lane; elem q -> M=q (+8 hi half).
    const int hi8 = (lane >> 4) * 8;
    #pragma unroll
    for (int j = 0; j < 2; ++j) {
        int n  = n_base + wn + j * 16 + r;
        float bv = bias[n];
        #pragma unroll
        for (int i = 0; i < 4; ++i) {
            int row0 = m_base + wm + i * 16 + hi8;
            #pragma unroll
            for (int q = 0; q < 8; ++q) {
                float v = acc[i][j][q] + bv;
                long o  = (long)(row0 + q) * N + n;
                if (MODE == 0) {
                    outb[o] = (__bf16)gelu_exact(v);
                } else {
                    float nv = out[o] + v;
                    out[o] = nv;
                    if (outb) outb[o] = (__bf16)nv;
                }
            }
        }
    }
}

// ---------------------------------------------------------------------------
// Host-side orchestration.
// ---------------------------------------------------------------------------
extern "C" void kernel_launch(void* const* d_in, const int* in_sizes, int n_in,
                              void* d_out, int out_size, void* d_ws, size_t ws_size,
                              hipStream_t stream) {
    (void)in_sizes; (void)n_in; (void)out_size; (void)ws_size;

    const float* x     = (const float*)d_in[0];
    const float* gamma = (const float*)d_in[1];
    const float* beta  = (const float*)d_in[2];
    const float* selw1[3]; const float* selb1[3]; const float* selw2[3]; const float* selb2[3];
    const float* linw1[3]; const float* linb1[3]; const float* linw2[3]; const float* linb2[3];
    for (int s = 0; s < 3; ++s) {
        int o = 3 + 8 * s;
        selw1[s] = (const float*)d_in[o + 0];
        selb1[s] = (const float*)d_in[o + 1];
        selw2[s] = (const float*)d_in[o + 2];
        selb2[s] = (const float*)d_in[o + 3];
        linw1[s] = (const float*)d_in[o + 4];
        linb1[s] = (const float*)d_in[o + 5];
        linw2[s] = (const float*)d_in[o + 6];
        linb2[s] = (const float*)d_in[o + 7];
    }
    float* out = (float*)d_out;

    // ---- workspace layout ----
    // f32 region
    float* ws = (float*)d_ws;
    float* xn    = ws;                       // 16,777,216
    float* avg0  = xn   + 16777216;          //  2,097,152   (-> sample0 f32)
    float* mx0   = avg0 + 2097152;           //  2,097,152
    float* avg1  = mx0  + 2097152;           //  4,194,304   (-> sample1 f32)
    float* mx1   = avg1 + 4194304;           //  4,194,304
    float* avg2  = mx1  + 4194304;           //  8,388,608   (-> sample2 f32)
    float* mx2   = avg2 + 8388608;           //  8,388,608
    float* stats = mx2  + 8388608;           //  3 * 64*384
    float* wsel  = stats + 3 * 64 * 384;     //  3 * 64
    // bf16 region (16B-aligned: f32 region is a multiple of 16B)
    __bf16* bregion = (__bf16*)(wsel + 3 * 64 + 32);
    __bf16* s0b  = bregion;                  //  2,097,152  sample0 bf16
    __bf16* s1b  = s0b + 2097152;            //  4,194,304  sample1 bf16
    __bf16* s2b  = s1b + 4194304;            //  8,388,608  sample2 bf16
    __bf16* Hb   = s2b + 8388608;            //  8,388,608  hidden bf16 (max)
    __bf16* wt1[3]; __bf16* wt2[3];
    {
        __bf16* p = Hb + 8388608;
        const int L1[3] = {256, 512, 1024};
        for (int s = 0; s < 3; ++s) {
            wt1[s] = p; p += L1[s] * L1[s];          // (L x L) transposed
            wt2[s] = p; p += L1[s] * 2 * L1[s];      // (2L x L) transposed
        }
    }

    const int kArr[3] = {8, 4, 2};
    const int LArr[3] = {SS / 8, SS / 4, SS / 2};     // 256, 512, 1024
    float* avgArr[3] = {avg0, avg1, avg2};
    float* mxArr[3]  = {mx0,  mx1,  mx2};
    __bf16* sbArr[3] = {s0b, s1b, s2b};

    // 0) one-time weight convert + transpose (independent of data path)
    for (int s = 0; s < 3; ++s) {
        int L = LArr[s];
        wprep_kernel<<<(L * L) / 256, 256, 0, stream>>>(linw1[s], wt1[s], L, L);
        wprep_kernel<<<(L * 2 * L) / 256, 256, 0, stream>>>(linw2[s], wt2[s], L, 2 * L);
    }

    // 1) BatchNorm (+ init d_out = xn)
    bn_kernel<<<FS / 256, 256, 0, stream>>>(x, gamma, beta, xn, out);

    // 2-5) pooling, stats, selector, mix per scale
    for (int s = 0; s < 3; ++s) {
        int L = LArr[s];
        pool_kernel<<<(MM * L) / 256, 256, 0, stream>>>(xn, avgArr[s], mxArr[s], L, kArr[s]);
        stats_kernel<<<MM, 256, 0, stream>>>(avgArr[s], mxArr[s], stats + s * 64 * 384, L);
        sel_kernel<<<BB, 64, 0, stream>>>(stats + s * 64 * 384, selw1[s], selb1[s],
                                          selw2[s], selb2[s], wsel + s * 64);
        mix_kernel<<<(MM * L) / 256, 256, 0, stream>>>(avgArr[s], mxArr[s], wsel + s * 64,
                                                       sbArr[s], L);
    }

    // 6) GEMM chain:
    //    Hb = bf16(gelu(sample_s @ W1_s + b1_s))
    //    sample_{s+1}(f32) += Hb @ W2_s + b2_s ; shadow bf16 for next A
    // s = 0
    gemm_kernel<0><<<dim3(2,  MM / 128), 256, 0, stream>>>(s0b, wt1[0], linb1[0], nullptr, Hb,  256,  256);
    gemm_kernel<1><<<dim3(4,  MM / 128), 256, 0, stream>>>(Hb,  wt2[0], linb2[0], avg1, s1b,   512,  256);
    // s = 1
    gemm_kernel<0><<<dim3(4,  MM / 128), 256, 0, stream>>>(s1b, wt1[1], linb1[1], nullptr, Hb,  512,  512);
    gemm_kernel<1><<<dim3(8,  MM / 128), 256, 0, stream>>>(Hb,  wt2[1], linb2[1], avg2, s2b,  1024,  512);
    // s = 2
    gemm_kernel<0><<<dim3(8,  MM / 128), 256, 0, stream>>>(s2b, wt1[2], linb1[2], nullptr, Hb, 1024, 1024);
    gemm_kernel<1><<<dim3(16, MM / 128), 256, 0, stream>>>(Hb,  wt2[2], linb2[2], out, nullptr, 2048, 1024);
}
